// PresoftmaxDifferentialOuterProductMean_22531398434972
// MI455X (gfx1250) — compile-verified, matrix operand order checked by reading, not run on hardware
//
#include <hip/hip_runtime.h>
#include <math.h>

// ---------------- types for WMMA ----------------
typedef __attribute__((ext_vector_type(16))) __bf16 v16bf;
typedef __attribute__((ext_vector_type(8)))  float  v8f;

union Frag {
    v16bf v;
    float4 f4[2];
};

// round-to-nearest-even f32 -> bf16 (raw u16)
static __device__ __forceinline__ unsigned short f2bf(float x) {
    unsigned u = __float_as_uint(x);
    unsigned r = (u + 0x7FFFu + ((u >> 16) & 1u)) >> 16;
    return (unsigned short)r;
}

// ---------------- constants ----------------
#define SB 1
#define SS 512
#define SN 384
#define SD 64
#define SH 32
#define SDP 128
#define SDQK 64
#define SCALING 0.125f   // 64^-0.5

// workspace layout (bytes)
#define OFF_Q    0                 // [2][384][64] f32      = 196608
#define OFF_SW   196608            // [2][512]     f32      = 4096
#define OFF_WSQ  200704            // [512]        f32      = 2048
#define OFF_WP   204800            // [256][1024]  bf16     = 524288
#define OFF_A    729088            // [12288][512] bf16     = 12582912
#define OFF_B    13312000          // [12288][512] bf16     = 12582912

// ---------------- helpers ----------------
static __device__ float blockReduce64(float v, volatile float* buf, int tid) {
    buf[tid] = v;
    __syncthreads();
    for (int off = 32; off > 0; off >>= 1) {
        if (tid < off) buf[tid] += buf[tid + off];
        __syncthreads();
    }
    float r = buf[0];
    __syncthreads();
    return r;
}

// ---------------- k1: q = LN_q( LN(msa[:,0]) @ Wq.T ) ----------------
__global__ __launch_bounds__(64) void k1_q(
    const float* __restrict__ msa, const float* __restrict__ ln_g,
    const float* __restrict__ ln_b, const float* __restrict__ Wq,
    const float* __restrict__ qn_g, const float* __restrict__ qn_b,
    float* __restrict__ qout)
{
    __shared__ float xs[64];
    __shared__ float red[64];
    const int n = blockIdx.x;
    const int d = threadIdx.x;

    float v   = msa[(0 * SN + n) * SD + d];          // s = 0
    float m   = blockReduce64(v, red, d) * (1.f / 64.f);
    float df  = v - m;
    float var = blockReduce64(df * df, red, d) * (1.f / 64.f);
    xs[d] = df * rsqrtf(var + 1e-5f) * ln_g[d] + ln_b[d];
    __syncthreads();

    for (int t = 0; t < 2; ++t) {
        float acc = 0.f;
        #pragma unroll
        for (int e = 0; e < 64; ++e)
            acc += Wq[(t * SDQK + d) * SD + e] * xs[e];
        float qm  = blockReduce64(acc, red, d) * (1.f / 64.f);
        float qd  = acc - qm;
        float qv  = blockReduce64(qd * qd, red, d) * (1.f / 64.f);
        qout[(t * SN + n) * SDQK + d] =
            qd * rsqrtf(qv + 1e-5f) * qn_g[d] + qn_b[d];
    }
}

// ---------------- k2: differential scores sw[t][s] ----------------
__global__ __launch_bounds__(64) void k2_scores(
    const float* __restrict__ msa, const float* __restrict__ ln_g,
    const float* __restrict__ ln_b, const float* __restrict__ Wk,
    const float* __restrict__ kn_g, const float* __restrict__ kn_b,
    const float* __restrict__ q, float* __restrict__ sw)
{
    __shared__ float xs[64];
    __shared__ float red[64];
    const int s = blockIdx.x;
    const int d = threadIdx.x;

    float acc0 = 0.f, acc1 = 0.f;
    for (int n = 0; n < SN; ++n) {
        float v   = msa[(s * SN + n) * SD + d];
        float m   = blockReduce64(v, red, d) * (1.f / 64.f);
        float df  = v - m;
        float var = blockReduce64(df * df, red, d) * (1.f / 64.f);
        xs[d] = df * rsqrtf(var + 1e-5f) * ln_g[d] + ln_b[d];
        __syncthreads();
        for (int t = 0; t < 2; ++t) {
            float k = 0.f;
            #pragma unroll
            for (int e = 0; e < 64; ++e)
                k += Wk[(t * SDQK + d) * SD + e] * xs[e];
            float km = blockReduce64(k, red, d) * (1.f / 64.f);
            float kd = k - km;
            float kv = blockReduce64(kd * kd, red, d) * (1.f / 64.f);
            float kn = kd * rsqrtf(kv + 1e-5f) * kn_g[d] + kn_b[d];
            float c  = q[(t * SN + n) * SDQK + d] * kn;
            if (t == 0) acc0 += c; else acc1 += c;
        }
        __syncthreads();
    }
    float t0 = blockReduce64(acc0, red, d);
    float t1 = blockReduce64(acc1, red, d);
    if (d == 0) {
        sw[s]      = t0 * SCALING / 384.f;
        sw[SS + s] = t1 * SCALING / 384.f;
    }
}

// ---------------- k3: lambda + softmax + sqrt ----------------
__global__ __launch_bounds__(512) void k3_softmax(
    const float* __restrict__ sw,
    const float* __restrict__ lq1, const float* __restrict__ lk1,
    const float* __restrict__ lq2, const float* __restrict__ lk2,
    float* __restrict__ wsq)
{
    __shared__ float red[512];
    const int t = threadIdx.x;
    float s1 = 0.f, s2 = 0.f;
    #pragma unroll
    for (int i = 0; i < 64; ++i) { s1 += lq1[i] * lk1[i]; s2 += lq2[i] * lk2[i]; }
    float lam = __expf(s1) - __expf(s2) + 1.0f;
    float v = sw[t] - lam * sw[SS + t];

    red[t] = v; __syncthreads();
    for (int off = 256; off > 0; off >>= 1) {
        if (t < off) red[t] = fmaxf(red[t], red[t + off]);
        __syncthreads();
    }
    float mx = red[0]; __syncthreads();
    float e = __expf(v - mx);
    red[t] = e; __syncthreads();
    for (int off = 256; off > 0; off >>= 1) {
        if (t < off) red[t] += red[t + off];
        __syncthreads();
    }
    float sum = red[0];
    wsq[t] = sqrtf(e / sum + 1e-32f);
}

// ---------------- k4: LN + Wl/Wr projections -> bf16 K-major bufs ----------------
__global__ __launch_bounds__(256) void k4_proj(
    const float* __restrict__ msa, const float* __restrict__ ln_g,
    const float* __restrict__ ln_b, const float* __restrict__ Wl,
    const float* __restrict__ Wr, const float* __restrict__ wsq,
    unsigned short* __restrict__ a_buf, unsigned short* __restrict__ b_buf)
{
    int g = blockIdx.x * 256 + threadIdx.x;  // 196608 = 512*384
    int s = g & 511;
    int n = g >> 9;
    const float* row = msa + (s * SN + n) * SD;

    float x[64];
    float sum = 0.f;
    #pragma unroll
    for (int d = 0; d < 64; ++d) { x[d] = row[d]; sum += x[d]; }
    float m = sum * (1.f / 64.f);
    float var = 0.f;
    #pragma unroll
    for (int d = 0; d < 64; ++d) { x[d] -= m; var += x[d] * x[d]; }
    float inv = rsqrtf(var * (1.f / 64.f) + 1e-5f);
    #pragma unroll
    for (int d = 0; d < 64; ++d) x[d] = x[d] * inv * ln_g[d] + ln_b[d];

    float w = wsq[s];
    for (int h = 0; h < SH; ++h) {
        float al = 0.f, bl = 0.f;
        #pragma unroll
        for (int d = 0; d < 64; ++d) {
            al += Wl[h * SD + d] * x[d];
            bl += Wr[h * SD + d] * x[d];
        }
        a_buf[(n * SH + h) * SS + s] = f2bf(al * w);
        b_buf[(n * SH + h) * SS + s] = f2bf(bl * w);
    }
}

// ---------------- k4b: Wp f32 -> bf16 ----------------
__global__ __launch_bounds__(256) void k4b_wp(
    const float* __restrict__ Wp, unsigned short* __restrict__ wp_bf)
{
    int i = blockIdx.x * 256 + threadIdx.x;
    if (i < 256 * 1024) wp_bf[i] = f2bf(Wp[i]);
}

// ---------------- k5: fused outer-product GEMM + Wp GEMM + SwiGLU ----------------
// grid (96, 96); block 256 (8 waves). Each WG handles a 4x4 (i,j) block.
__global__ __launch_bounds__(256) void k5_fused(
    const unsigned short* __restrict__ a_buf,  // [12288][512] bf16
    const unsigned short* __restrict__ b_buf,  // [12288][512] bf16
    const unsigned short* __restrict__ wp_bf,  // [256][1024]  bf16
    const float* __restrict__ bp,              // [256]
    float* __restrict__ out)                   // [384][384][128]
{
    __shared__ unsigned short lds_outer[16 * 1024];  // 32 KB; reused as f32[16][256]

    const int tid   = threadIdx.x;
    const int wave  = tid >> 5;
    const int lane  = tid & 31;
    const int lhalf = lane & 15;
    const int lhi   = lane >> 4;
    const int ib = blockIdx.x;
    const int jb = blockIdx.y;
    const int M0 = ib * 128;
    const int N0 = jb * 128;

    // ---- stage 1: 128x128x512 bf16 WMMA GEMM (outer-product-mean tile) ----
    const int wm = wave >> 2;   // 0..1  -> 64 rows
    const int wn = wave & 3;    // 0..3  -> 32 cols
    v8f acc[4][2];
    #pragma unroll
    for (int i = 0; i < 4; ++i)
        #pragma unroll
        for (int j = 0; j < 2; ++j)
            acc[i][j] = v8f{};

    for (int ks = 0; ks < 16; ++ks) {
        const int kbase = ks * 32 + lhi * 8;
        Frag afr[4], bfr[2];
        #pragma unroll
        for (int mt = 0; mt < 4; ++mt) {
            const unsigned short* p =
                a_buf + (size_t)(M0 + wm * 64 + mt * 16 + lhalf) * SS + kbase;
            afr[mt].f4[0] = *(const float4*)(p);
            afr[mt].f4[1] = *(const float4*)(p + 16);
        }
        #pragma unroll
        for (int nt = 0; nt < 2; ++nt) {
            const unsigned short* p =
                b_buf + (size_t)(N0 + wn * 32 + nt * 16 + lhalf) * SS + kbase;
            bfr[nt].f4[0] = *(const float4*)(p);
            bfr[nt].f4[1] = *(const float4*)(p + 16);
        }
        #pragma unroll
        for (int mt = 0; mt < 4; ++mt)
            #pragma unroll
            for (int nt = 0; nt < 2; ++nt)
                acc[mt][nt] = __builtin_amdgcn_wmma_f32_16x16x32_bf16(
                    false, afr[mt].v, false, bfr[nt].v,
                    (short)0, acc[mt][nt], false, false);
    }

    // scatter f32 accums -> bf16 LDS [pair][c*32+e], pair = li*4+lj
    #pragma unroll
    for (int mt = 0; mt < 4; ++mt) {
        #pragma unroll
        for (int nt = 0; nt < 2; ++nt) {
            #pragma unroll
            for (int r = 0; r < 8; ++r) {
                int m = wm * 64 + mt * 16 + r + 8 * lhi;  // 0..127
                int n = wn * 32 + nt * 16 + lhalf;        // 0..127
                int li = m >> 5, c = m & 31;
                int lj = n >> 5, e = n & 31;
                lds_outer[((li * 4 + lj) << 10) + c * 32 + e] = f2bf(acc[mt][nt][r]);
            }
        }
    }
    __syncthreads();

    // ---- stage 2: [16 pairs] x [256] x K=1024 bf16 WMMA GEMM against Wp ----
    v8f acc2[2];
    acc2[0] = v8f{};
    acc2[1] = v8f{};
    for (int ks = 0; ks < 32; ++ks) {
        const int kbase = ks * 32 + lhi * 8;
        Frag a2;
        const unsigned short* p = lds_outer + (lhalf << 10) + kbase;
        a2.f4[0] = *(const float4*)(p);
        a2.f4[1] = *(const float4*)(p + 16);
        #pragma unroll
        for (int nt = 0; nt < 2; ++nt) {
            const unsigned short* qp =
                wp_bf + (size_t)(wave * 32 + nt * 16 + lhalf) * 1024 + kbase;
            Frag b2;
            b2.f4[0] = *(const float4*)(qp);
            b2.f4[1] = *(const float4*)(qp + 16);
            acc2[nt] = __builtin_amdgcn_wmma_f32_16x16x32_bf16(
                false, a2.v, false, b2.v, (short)0, acc2[nt], false, false);
        }
    }
    __syncthreads();   // done reading lds_outer; reuse LDS as f32 [16][256]

    float* lds_out = (float*)lds_outer;
    #pragma unroll
    for (int nt = 0; nt < 2; ++nt) {
        #pragma unroll
        for (int r = 0; r < 8; ++r) {
            int m = r + 8 * lhi;                    // pair 0..15
            int p = wave * 32 + nt * 16 + lhalf;    // 0..255
            lds_out[m * 256 + p] = acc2[nt][r] + bp[p];
        }
    }
    __syncthreads();

    // SwiGLU + store: out[i][j][d] = silu(gate) * val
    for (int idx = tid; idx < 16 * SDP; idx += 256) {
        int pair = idx >> 7;
        int d    = idx & 127;
        float val = lds_out[pair * 256 + d];
        float g   = lds_out[pair * 256 + SDP + d];
        float sig = 1.f / (1.f + __expf(-g));
        int li = pair >> 2, lj = pair & 3;
        int gi = ib * 4 + li, gj = jb * 4 + lj;
        out[((size_t)gi * SN + gj) * SDP + d] = g * sig * val;
    }
}

// ---------------- launch ----------------
extern "C" void kernel_launch(void* const* d_in, const int* in_sizes, int n_in,
                              void* d_out, int out_size, void* d_ws, size_t ws_size,
                              hipStream_t stream) {
    const float* msa  = (const float*)d_in[0];
    const float* ln_g = (const float*)d_in[1];
    const float* ln_b = (const float*)d_in[2];
    const float* Wq   = (const float*)d_in[3];
    const float* Wk   = (const float*)d_in[4];
    const float* qn_g = (const float*)d_in[5];
    const float* qn_b = (const float*)d_in[6];
    const float* kn_g = (const float*)d_in[7];
    const float* kn_b = (const float*)d_in[8];
    const float* lq1  = (const float*)d_in[9];
    const float* lk1  = (const float*)d_in[10];
    const float* lq2  = (const float*)d_in[11];
    const float* lk2  = (const float*)d_in[12];
    const float* Wl   = (const float*)d_in[13];
    const float* Wr   = (const float*)d_in[14];
    const float* Wp   = (const float*)d_in[15];
    const float* bp   = (const float*)d_in[16];

    char* ws = (char*)d_ws;
    float*          q_buf  = (float*)(ws + OFF_Q);
    float*          sw     = (float*)(ws + OFF_SW);
    float*          wsq    = (float*)(ws + OFF_WSQ);
    unsigned short* wp_bf  = (unsigned short*)(ws + OFF_WP);
    unsigned short* a_buf  = (unsigned short*)(ws + OFF_A);
    unsigned short* b_buf  = (unsigned short*)(ws + OFF_B);

    k1_q      <<<SN, 64, 0, stream>>>(msa, ln_g, ln_b, Wq, qn_g, qn_b, q_buf);
    k2_scores <<<SS, 64, 0, stream>>>(msa, ln_g, ln_b, Wk, kn_g, kn_b, q_buf, sw);
    k3_softmax<<<1, 512, 0, stream>>>(sw, lq1, lk1, lq2, lk2, wsq);
    k4_proj   <<<(SS * SN) / 256, 256, 0, stream>>>(msa, ln_g, ln_b, Wl, Wr, wsq,
                                                    a_buf, b_buf);
    k4b_wp    <<<1024, 256, 0, stream>>>(Wp, wp_bf);
    k5_fused  <<<dim3(SN / 4, SN / 4), 256, 0, stream>>>(a_buf, b_buf, wp_bf, bp,
                                                         (float*)d_out);
}
